// TACrossAttention_11476152615422
// MI455X (gfx1250) — matrix operand order repaired and checked by reading, not compiled, not run on hardware
//
#include <hip/hip_runtime.h>

typedef __bf16 bf16;
typedef __attribute__((ext_vector_type(16))) __bf16 v16bf;
typedef __attribute__((ext_vector_type(8)))  __bf16 v8bf;
typedef __attribute__((ext_vector_type(8)))  float  v8f;
typedef __attribute__((ext_vector_type(4)))  int    v4i;

union V16 { v16bf v; v8bf h[2]; };

#define Bc     2
#define TQc    2048
#define TCc    2048
#define DQc    1024
#define DCc    768
#define Hc     12
#define DHc    64
#define INNERc 768

__device__ __forceinline__ bf16 f2bf(float f) {
  union { float f; unsigned u; } x; x.f = f;
  unsigned r = x.u + 0x7FFFu + ((x.u >> 16) & 1u);
  union { unsigned short s; bf16 b; } y; y.s = (unsigned short)(r >> 16);
  return y.b;
}

__device__ __forceinline__ v8bf ld8(const bf16* p) { return *(const v8bf*)p; }

// async global->LDS 16B copy (GLOBAL_LOAD_ASYNC_TO_LDS_B128, ASYNCcnt-tracked)
__device__ __forceinline__ void async_cp16(const bf16* g, bf16* l) {
  __builtin_amdgcn_global_load_async_to_lds_b128(
      (__attribute__((address_space(1))) v4i*)g,
      (__attribute__((address_space(3))) v4i*)l, 0, 0);
}

// ---------------- f32 -> bf16 conversion ----------------
__global__ void cvt_bf16(const float* __restrict__ s, bf16* __restrict__ d, int n) {
  int i = blockIdx.x * blockDim.x + threadIdx.x;
  if (i < n) d[i] = f2bf(s[i]);
}

// ---------------- GEMM: C[M,N] = A[M,K] * W[N,K]^T + bias ----------------
// mode 0: bf16 row-major out, scaled by alpha
// mode 1: bf16 transposed-per-batch out: out[b][n][c], b=m/trows, c=m%trows
// mode 2: f32 row-major out
__global__ __launch_bounds__(256) void gemm_bf16_wmma(
    const bf16* __restrict__ A, const bf16* __restrict__ W,
    const float* __restrict__ bias, void* __restrict__ Cout,
    int M, int N, int Kd, float alpha, int mode, int trows) {
  int lane = threadIdx.x & 31;
  int wave = threadIdx.x >> 5;
  int wm = wave >> 1, wn = wave & 1;
  int m0 = blockIdx.x * 128 + wm * 32;
  int n0 = blockIdx.y * 64 + wn * 32;
  int lo = lane & 15, hi = lane >> 4;

  v8f acc[2][2] = {};
  const bf16* arow[2];
  const bf16* wrow[2];
  arow[0] = A + (size_t)(m0 + lo) * Kd;
  arow[1] = A + (size_t)(m0 + 16 + lo) * Kd;
  wrow[0] = W + (size_t)(n0 + lo) * Kd;
  wrow[1] = W + (size_t)(n0 + 16 + lo) * Kd;

  for (int k = 0; k < Kd; k += 32) {
    V16 af[2], bfr[2];
#pragma unroll
    for (int mt = 0; mt < 2; ++mt) {
      const bf16* p = arow[mt] + k + hi * 8;   // A-frag: K = hi*8..+7 and 16+hi*8..+7
      af[mt].h[0] = ld8(p);
      af[mt].h[1] = ld8(p + 16);
    }
#pragma unroll
    for (int nt = 0; nt < 2; ++nt) {
      const bf16* p = wrow[nt] + k + hi * 8;   // B-frag: K = hi*8..+7 and 16+hi*8..+7
      bfr[nt].h[0] = ld8(p);
      bfr[nt].h[1] = ld8(p + 16);
    }
#pragma unroll
    for (int mt = 0; mt < 2; ++mt)
#pragma unroll
      for (int nt = 0; nt < 2; ++nt)
        acc[mt][nt] = __builtin_amdgcn_wmma_f32_16x16x32_bf16(
            false, af[mt].v, false, bfr[nt].v, (short)0, acc[mt][nt], false, false);
  }

#pragma unroll
  for (int mt = 0; mt < 2; ++mt) {
#pragma unroll
    for (int nt = 0; nt < 2; ++nt) {
      int n = n0 + nt * 16 + lo;
      float bv = bias ? bias[n] : 0.f;
#pragma unroll
      for (int r = 0; r < 8; ++r) {
        int m = m0 + mt * 16 + 8 * hi + r;
        float v = (acc[mt][nt][r] + bv) * alpha;
        if (mode == 0) {
          ((bf16*)Cout)[(size_t)m * N + n] = f2bf(v);
        } else if (mode == 1) {
          int bidx = m / trows, c = m % trows;
          ((bf16*)Cout)[((size_t)bidx * N + n) * trows + c] = f2bf(v);
        } else {
          ((float*)Cout)[(size_t)m * N + n] = v;
        }
      }
    }
  }
}

// stage one 32-key chunk of K (32x64) and V^T (64x32) into LDS: 4 async b128/thread
__device__ __forceinline__ void stage_chunk(
    const bf16* __restrict__ K, const bf16* __restrict__ Vt,
    int b, int h, int c0, bf16* kbuf, bf16* vbuf, int t) {
#pragma unroll
  for (int i = 0; i < 2; ++i) {
    int c = t + i * 128;             // 256 x 16B chunks
    int row = c >> 3, col = (c & 7) * 8;
    async_cp16(K + (size_t)(b * TCc + c0 + row) * INNERc + h * DHc + col,
               kbuf + row * 64 + col);
  }
#pragma unroll
  for (int i = 0; i < 2; ++i) {
    int c = t + i * 128;
    int row = c >> 2, col = (c & 3) * 8;
    async_cp16(Vt + (size_t)(b * INNERc + h * DHc + row) * TCc + c0 + col,
               vbuf + row * 32 + col);
  }
}

// ---------------- Flash attention (block = 4 waves = 128 q-rows of one head) ----------------
__global__ __launch_bounds__(128) void flash_attn(
    const bf16* __restrict__ Q, const bf16* __restrict__ K,
    const bf16* __restrict__ Vt, const int* __restrict__ mask,
    bf16* __restrict__ O) {
  int t = threadIdx.x;
  int lane = t & 31;
  int wave = t >> 5;
  int lo = lane & 15, hi = lane >> 4;
  const int NQB = TQc / 128;
  int qblk = blockIdx.x % NQB;
  int bh = blockIdx.x / NQB;
  int b = bh / Hc, h = bh % Hc;
  int q0 = qblk * 128 + wave * 32;

  __shared__ __align__(16) bf16 kbuf[2][32 * 64];   // K chunk, double buffered
  __shared__ __align__(16) bf16 vbuf[2][64 * 32];   // V^T chunk, double buffered
  __shared__ __align__(16) bf16 ldsP[4][2][16 * 40]; // per-wave P tiles

  // Q fragments: resident for whole kernel (pre-scaled by 1/8)
  V16 qf[2][2];
#pragma unroll
  for (int mt = 0; mt < 2; ++mt)
#pragma unroll
    for (int ks = 0; ks < 2; ++ks) {
      const bf16* p = Q + (size_t)(b * TQc + q0 + mt * 16 + lo) * INNERc + h * DHc + ks * 32 + hi * 8;
      qf[mt][ks].h[0] = ld8(p);
      qf[mt][ks].h[1] = ld8(p + 16);
    }

  v8f o[2][4] = {};
  float rmax[2][8], rsum[2][8];
#pragma unroll
  for (int mt = 0; mt < 2; ++mt)
#pragma unroll
    for (int r = 0; r < 8; ++r) { rmax[mt][r] = -1e30f; rsum[mt][r] = 0.f; }

  // prime the pipeline
  stage_chunk(K, Vt, b, h, 0, kbuf[0], vbuf[0], t);

  for (int c0 = 0; c0 < TCc; c0 += 32) {
    int cur = (c0 >> 5) & 1;
    if (c0 + 32 < TCc) {
      stage_chunk(K, Vt, b, h, c0 + 32, kbuf[cur ^ 1], vbuf[cur ^ 1], t);
      __builtin_amdgcn_s_wait_asynccnt(4);  // current chunk's 4 copies done
    } else {
      __builtin_amdgcn_s_wait_asynccnt(0);
    }
    __syncthreads();  // all waves' copies for this chunk visible

    // K B-fragments from LDS
    V16 kf[2][2];
#pragma unroll
    for (int nt = 0; nt < 2; ++nt)
#pragma unroll
      for (int ks = 0; ks < 2; ++ks) {
        const bf16* p = kbuf[cur] + (nt * 16 + lo) * 64 + ks * 32 + hi * 8;
        kf[nt][ks].h[0] = *(const v8bf*)p;
        kf[nt][ks].h[1] = *(const v8bf*)(p + 16);
      }

    // key-padding mask: C-layout column = this lane's lo
    float mm[2];
#pragma unroll
    for (int nt = 0; nt < 2; ++nt)
      mm[nt] = mask[b * TCc + c0 + nt * 16 + lo] ? 0.f : 1.f;

    // S = Q K^T (pre-scaled)
    v8f s[2][2];
#pragma unroll
    for (int mt = 0; mt < 2; ++mt)
#pragma unroll
      for (int nt = 0; nt < 2; ++nt) {
        v8f z = {};
        z = __builtin_amdgcn_wmma_f32_16x16x32_bf16(false, qf[mt][0].v, false, kf[nt][0].v, (short)0, z, false, false);
        z = __builtin_amdgcn_wmma_f32_16x16x32_bf16(false, qf[mt][1].v, false, kf[nt][1].v, (short)0, z, false, false);
        s[mt][nt] = z;
      }

    // masked -> -1e30 sentinel
#pragma unroll
    for (int mt = 0; mt < 2; ++mt)
#pragma unroll
      for (int nt = 0; nt < 2; ++nt)
#pragma unroll
        for (int r = 0; r < 8; ++r)
          s[mt][nt][r] = s[mt][nt][r] * mm[nt] + (mm[nt] - 1.f) * 1e30f;

    // online softmax per 16-row tile
#pragma unroll
    for (int mt = 0; mt < 2; ++mt) {
      float sc[8];
#pragma unroll
      for (int r = 0; r < 8; ++r) {
        float tm = fmaxf(s[mt][0][r], s[mt][1][r]);
#pragma unroll
        for (int off = 1; off < 16; off <<= 1)
          tm = fmaxf(tm, __shfl_xor(tm, off, 32));
        float nm = fmaxf(rmax[mt][r], tm);
        sc[r] = __expf(rmax[mt][r] - nm);
        rmax[mt][r] = nm;
      }
#pragma unroll
      for (int r = 0; r < 8; ++r) {
        float p0 = mm[0] * __expf(s[mt][0][r] - rmax[mt][r]);
        float p1 = mm[1] * __expf(s[mt][1][r] - rmax[mt][r]);
        int row = 8 * hi + r;
        ldsP[wave][mt][row * 40 + lo] = f2bf(p0);
        ldsP[wave][mt][row * 40 + 16 + lo] = f2bf(p1);
        float ts = p0 + p1;
#pragma unroll
        for (int off = 1; off < 16; off <<= 1)
          ts += __shfl_xor(ts, off, 32);
        rsum[mt][r] = rsum[mt][r] * sc[r] + ts;
      }
#pragma unroll
      for (int dt = 0; dt < 4; ++dt)
#pragma unroll
        for (int r = 0; r < 8; ++r)
          o[mt][dt][r] *= sc[r];
    }

    // P A-fragments (C-layout -> A-layout via LDS)
    V16 pf[2];
#pragma unroll
    for (int mt = 0; mt < 2; ++mt) {
      const bf16* base = &ldsP[wave][mt][0];
      pf[mt].h[0] = *(const v8bf*)(base + lo * 40 + hi * 8);
      pf[mt].h[1] = *(const v8bf*)(base + lo * 40 + 16 + hi * 8);
    }

    // V^T B-fragments from LDS: row = dim, contiguous keys
    V16 vf[4];
#pragma unroll
    for (int dt = 0; dt < 4; ++dt) {
      const bf16* p = vbuf[cur] + (dt * 16 + lo) * 32 + hi * 8;
      vf[dt].h[0] = *(const v8bf*)p;
      vf[dt].h[1] = *(const v8bf*)(p + 16);
    }
#pragma unroll
    for (int mt = 0; mt < 2; ++mt)
#pragma unroll
      for (int dt = 0; dt < 4; ++dt)
        o[mt][dt] = __builtin_amdgcn_wmma_f32_16x16x32_bf16(
            false, pf[mt].v, false, vf[dt].v, (short)0, o[mt][dt], false, false);

    __syncthreads();  // done reading buffers before next overwrite
  }

  // finalize: divide by softmax denom, write bf16 row-major [B*TQ, INNER]
#pragma unroll
  for (int mt = 0; mt < 2; ++mt) {
    float inv[8];
#pragma unroll
    for (int r = 0; r < 8; ++r) inv[r] = 1.f / rsum[mt][r];
#pragma unroll
    for (int dt = 0; dt < 4; ++dt)
#pragma unroll
      for (int r = 0; r < 8; ++r) {
        int q = q0 + mt * 16 + 8 * hi + r;
        O[(size_t)(b * TQc + q) * INNERc + h * DHc + dt * 16 + lo] = f2bf(o[mt][dt][r] * inv[r]);
      }
  }
}

extern "C" void kernel_launch(void* const* d_in, const int* in_sizes, int n_in,
                              void* d_out, int out_size, void* d_ws, size_t ws_size,
                              hipStream_t stream) {
  (void)in_sizes; (void)n_in; (void)out_size; (void)ws_size;
  const float* query = (const float*)d_in[0];
  const float* cross = (const float*)d_in[1];
  const int*   mask  = (const int*)d_in[2];
  const float* Wq = (const float*)d_in[3];
  const float* bq = (const float*)d_in[4];
  const float* Wk = (const float*)d_in[5];
  const float* bk = (const float*)d_in[6];
  const float* Wv = (const float*)d_in[7];
  const float* bv = (const float*)d_in[8];
  const float* Wo = (const float*)d_in[9];
  const float* bo = (const float*)d_in[10];
  float* out = (float*)d_out;

  const int M = Bc * TQc;  // 4096

  char* ws = (char*)d_ws;
  size_t off = 0;
  auto alloc = [&](size_t bytes) -> void* {
    void* p = ws + off;
    off += bytes;
    off = (off + 255) & ~(size_t)255;
    return p;
  };
  bf16* qbf = (bf16*)alloc((size_t)M * DQc * 2);           // query bf16
  bf16* cbf = (bf16*)alloc((size_t)M * DCc * 2);           // cross bf16
  bf16* wqb = (bf16*)alloc((size_t)INNERc * DQc * 2);
  bf16* wkb = (bf16*)alloc((size_t)INNERc * DCc * 2);
  bf16* wvb = (bf16*)alloc((size_t)INNERc * DCc * 2);
  bf16* wob = (bf16*)alloc((size_t)DQc * INNERc * 2);
  bf16* Qp  = (bf16*)alloc((size_t)M * INNERc * 2);        // Q projection (pre-scaled)
  bf16* Kp  = (bf16*)alloc((size_t)M * INNERc * 2);        // K projection
  bf16* Vtp = (bf16*)alloc((size_t)Bc * INNERc * TCc * 2); // V projection, transposed
  bf16* AO  = (bf16*)alloc((size_t)M * INNERc * 2);        // attention output

  auto cvt = [&](const float* s, bf16* d, int n) {
    cvt_bf16<<<(n + 255) / 256, 256, 0, stream>>>(s, d, n);
  };
  cvt(query, qbf, M * DQc);
  cvt(cross, cbf, M * DCc);
  cvt(Wq, wqb, INNERc * DQc);
  cvt(Wk, wkb, INNERc * DCc);
  cvt(Wv, wvb, INNERc * DCc);
  cvt(Wo, wob, DQc * INNERc);

  // Q = (query Wq^T + bq) * (1/sqrt(64))
  gemm_bf16_wmma<<<dim3(M / 128, INNERc / 64), 256, 0, stream>>>(
      qbf, wqb, bq, Qp, M, INNERc, DQc, 0.125f, 0, 0);
  // K = cross Wk^T + bk
  gemm_bf16_wmma<<<dim3(M / 128, INNERc / 64), 256, 0, stream>>>(
      cbf, wkb, bk, Kp, M, INNERc, DCc, 1.0f, 0, 0);
  // V = cross Wv^T + bv, stored transposed per batch: Vt[b][d][c]
  gemm_bf16_wmma<<<dim3(M / 128, INNERc / 64), 256, 0, stream>>>(
      cbf, wvb, bv, Vtp, M, INNERc, DCc, 1.0f, 1, TCc);

  // fused masked softmax attention with async K/V staging
  flash_attn<<<Bc * Hc * (TQc / 128), 128, 0, stream>>>(Qp, Kp, Vtp, mask, AO);

  // out = AO Wo^T + bo (f32)
  gemm_bf16_wmma<<<dim3(M / 128, DQc / 64), 256, 0, stream>>>(
      AO, wob, bo, out, M, DQc, INNERc, 1.0f, 2, 0);
}
